// SelfAttention_41893111005418
// MI455X (gfx1250) — matrix operand order repaired and checked by reading, not compiled, not run on hardware
//
#include <hip/hip_runtime.h>

typedef unsigned short u16;
typedef __attribute__((ext_vector_type(16))) __bf16 v16bf;
typedef __attribute__((ext_vector_type(8)))  __bf16 v8bf;
typedef __attribute__((ext_vector_type(8)))  float  v8f;
typedef __attribute__((ext_vector_type(4))) unsigned int u32x4;
typedef __attribute__((ext_vector_type(8))) int i32x8;
typedef __attribute__((ext_vector_type(4))) int i32x4;

#if defined(__has_builtin)
#if __has_builtin(__builtin_amdgcn_tensor_load_to_lds) && __has_builtin(__builtin_amdgcn_s_wait_tensorcnt)
#define HAVE_TDM 1
#endif
#endif

#define BB 4
#define LL 2048
#define CC 1024
#define HH 16
#define DD 64
#define MM (BB*LL)   // 8192
#define NN (3*CC)    // 3072
#define KK CC        // 1024

__device__ __forceinline__ u16 f2bf(float f) {
  union { float f; unsigned u; } x; x.f = f;
  unsigned u = x.u;
  u += 0x7fffu + ((u >> 16) & 1u);   // round to nearest even
  return (u16)(u >> 16);
}

__device__ __forceinline__ v16bf cat16(v8bf lo, v8bf hi) {
  return __builtin_shufflevector(lo, hi, 0,1,2,3,4,5,6,7,8,9,10,11,12,13,14,15);
}

__device__ __forceinline__ v8f v8f_zero() {
  v8f z;
  #pragma unroll
  for (int i = 0; i < 8; ++i) z[i] = 0.0f;
  return z;
}

// ---------------- f32 -> bf16 conversion of x ----------------
__global__ __launch_bounds__(256) void cvt_x_kernel(const float4* __restrict__ x,
                                                    u16* __restrict__ xb, int n4) {
  int i = blockIdx.x * 256 + threadIdx.x;
  if (i >= n4) return;
  float4 v = x[i];
  u16* p = xb + (size_t)i * 4;
  p[0] = f2bf(v.x); p[1] = f2bf(v.y); p[2] = f2bf(v.z); p[3] = f2bf(v.w);
}

// ---------------- f32 W [K][N] -> bf16 Wt [N][K] ----------------
__global__ __launch_bounds__(256) void cvt_wT_kernel(const float* __restrict__ W,
                                                     u16* __restrict__ wt) {
  int i = blockIdx.x * 256 + threadIdx.x;
  if (i >= KK * NN) return;
  int n = i % NN, k = i / NN;                 // coalesced read of W row
  wt[(size_t)n * KK + k] = f2bf(W[i]);
}

// ---------------- QKV GEMM: [M,K] x [K,N] + bias -> Q/K/V bf16 [B,H,L,D] ----------------
__global__ __launch_bounds__(256) void qkv_gemm_kernel(
    const u16* __restrict__ xb, const u16* __restrict__ wt,
    const float* __restrict__ bias,
    u16* __restrict__ qw, u16* __restrict__ kw, u16* __restrict__ vw) {
  const int tid   = threadIdx.x;
  const int lane  = tid & 31;
  const int wid   = tid >> 5;
  const int wm    = wid & 3;              // 4 waves along M
  const int wn    = wid >> 2;             // 2 waves along N
  const int bm    = blockIdx.y * 128;
  const int bn    = blockIdx.x * 128;
  const int nlane = lane & 15;
  const int hiA   = (lane >> 4) << 3;     // A frag: K base 0 / 8
  const int hiB   = (lane >> 4) << 3;     // B frag: K base 0 / 8
  const int hi8   = (lane >> 4) << 3;     // C frag: row offset 0 / 8

  v8f acc[2][4];
  #pragma unroll
  for (int fm = 0; fm < 2; ++fm)
    #pragma unroll
    for (int fn = 0; fn < 4; ++fn) acc[fm][fn] = v8f_zero();

  for (int kk = 0; kk < KK; kk += 32) {
    v16bf a[2];
    #pragma unroll
    for (int fm = 0; fm < 2; ++fm) {
      int row = bm + wm * 32 + fm * 16 + nlane;
      const u16* p = xb + (size_t)row * KK + kk + hiA;
      a[fm] = cat16(*(const v8bf*)p, *(const v8bf*)(p + 16));
    }
    #pragma unroll
    for (int fn = 0; fn < 4; ++fn) {
      int col = bn + wn * 64 + fn * 16 + nlane;
      const u16* p = wt + (size_t)col * KK + kk + hiB;
      v16bf b = cat16(*(const v8bf*)p, *(const v8bf*)(p + 16));
      #pragma unroll
      for (int fm = 0; fm < 2; ++fm)
        acc[fm][fn] = __builtin_amdgcn_wmma_f32_16x16x32_bf16(
            false, a[fm], false, b, (short)0, acc[fm][fn], false, false);
    }
  }

  // epilogue: add bias, scatter to Q/K/V in [B,H,L,D] bf16
  #pragma unroll
  for (int fn = 0; fn < 4; ++fn) {
    int ncol = bn + wn * 64 + fn * 16 + nlane;
    float bv = bias[ncol];
    int which = ncol >> 10;               // 0=Q 1=K 2=V
    int cc2 = ncol & 1023;
    int hh = cc2 >> 6, dd = cc2 & 63;
    u16* dst = (which == 0) ? qw : (which == 1) ? kw : vw;
    #pragma unroll
    for (int fm = 0; fm < 2; ++fm) {
      #pragma unroll
      for (int r = 0; r < 8; ++r) {
        int mrow = bm + wm * 32 + fm * 16 + r + hi8;
        int bi = mrow >> 11, li = mrow & 2047;
        size_t off = (((size_t)(bi * HH + hh) * LL + li) << 6) + dd;
        dst[off] = f2bf(acc[fm][fn][r] + bv);
      }
    }
  }
}

// ---------------- Flash attention: 128 query rows per block, 64-wide KV blocks ----------------
__global__ __launch_bounds__(256) void attn_kernel(
    const u16* __restrict__ Q, const u16* __restrict__ Kmat, const u16* __restrict__ V,
    float* __restrict__ out) {
  __shared__ u16 Ks[64 * 64];         // K block,  [seq][d] == Bt layout for Q*K^T
  __shared__ u16 Vt[64 * 64];         // V block transposed, [d][seq] == Bt layout for P*V
  __shared__ u16 Pb[8 * 16 * 64];     // per-wave P staging (C-layout f32 -> A-layout bf16)

  const int tid   = threadIdx.x;
  const int lane  = tid & 31;
  const int wid   = tid >> 5;
  const int nlane = lane & 15;
  const int hiA   = (lane >> 4) << 3;
  const int hiB   = (lane >> 4) << 3;
  const int hi8   = hiA;

  const int ib = blockIdx.x;          // query row block (0..15)
  const int h  = blockIdx.y;
  const int bi = blockIdx.z;
  const int rbase = ib * 128 + wid * 16;

  const size_t headoff = (size_t)(bi * HH + h) * LL * DD;
  const u16* Qg = Q + headoff;
  const u16* Kg = Kmat + headoff;
  const u16* Vg = V + headoff;

  // Q fragments for this wave's 16 rows (K = 0..63 in two 32 slices)
  v16bf qf[2];
  #pragma unroll
  for (int kkk = 0; kkk < 2; ++kkk) {
    const u16* p = Qg + (size_t)(rbase + nlane) * DD + kkk * 32 + hiA;
    qf[kkk] = cat16(*(const v8bf*)p, *(const v8bf*)(p + 16));
  }

  v8f o[4];
  #pragma unroll
  for (int fd = 0; fd < 4; ++fd) o[fd] = v8f_zero();
  float m_r[8], l_r[8];
  #pragma unroll
  for (int r = 0; r < 8; ++r) { m_r[r] = -1e30f; l_r[r] = 0.0f; }

  u16* Pw = Pb + wid * 1024;
  const float scale = 0.125f;         // 1/sqrt(64)
  const int numj = 2 * ib + 2;        // causal: only KV blocks intersecting the triangle

  for (int jb = 0; jb < numj; ++jb) {
    __syncthreads();                  // previous iteration's LDS reads done

#ifdef HAVE_TDM
    // K block 64x64 bf16: single TDM descriptor, issued by wave 0, lands in Ks.
    if (wid == 0) {
      unsigned long long ga =
          (unsigned long long)(size_t)(const void*)(Kg + (size_t)jb * 64 * DD);
      unsigned ldsa = (unsigned)(size_t)(void*)&Ks[0];
      u32x4 g0;
      g0[0] = 1u;                                    // count=1, user descriptor
      g0[1] = ldsa;                                  // lds_addr
      g0[2] = (unsigned)(ga & 0xffffffffu);          // global_addr[31:0]
      g0[3] = (unsigned)((ga >> 32) & 0x01ffffffu)   // global_addr[56:32]
              | (2u << 30);                          // type=2 ("image")
      i32x8 g1;
      g1[0] = (int)(1u << 16);                       // data_size=1 -> 2 bytes/elem
      g1[1] = (int)(64u << 16);                      // tensor_dim0 = 64
      g1[2] = (int)((unsigned)(LL - jb * 64) << 16); // tensor_dim1 = remaining rows
      g1[3] = (int)(64u << 16);                      // tile_dim0 = 64
      g1[4] = 64;                                    // tile_dim1 = 64
      g1[5] = 64;                                    // tensor_dim0_stride = 64
      g1[6] = 0;
      g1[7] = 0;
      i32x4 gz4;
      gz4[0] = 0; gz4[1] = 0; gz4[2] = 0; gz4[3] = 0; // groups 2/3 unused (2-D tensor)
      i32x8 gz8;
      #pragma unroll
      for (int e = 0; e < 8; ++e) gz8[e] = 0;
      __builtin_amdgcn_tensor_load_to_lds(g0, g1, gz4, gz4, gz8, 0);
      __builtin_amdgcn_s_wait_tensorcnt(0);
    }
#else
    #pragma unroll
    for (int it = 0; it < 2; ++it) {
      int c = tid + it * 256;
      int s = c >> 3, seg = c & 7;
      *(v8bf*)&Ks[s * 64 + seg * 8] =
          *(const v8bf*)(Kg + (size_t)(jb * 64 + s) * DD + seg * 8);
    }
#endif
    // V block -> Vt (transposed through registers)
    #pragma unroll
    for (int it = 0; it < 2; ++it) {
      int c = tid + it * 256;
      int s = c >> 3, seg = c & 7;
      union { v8bf v; u16 u[8]; } t;
      t.v = *(const v8bf*)(Vg + (size_t)(jb * 64 + s) * DD + seg * 8);
      #pragma unroll
      for (int e = 0; e < 8; ++e) Vt[(seg * 8 + e) * 64 + s] = t.u[e];
    }
    __syncthreads();

    if (jb * 64 > rbase + 15) continue;   // wave-uniform: block fully masked

    // S = Q * K^T (16 x 64)
    v8f s[4];
    #pragma unroll
    for (int fn = 0; fn < 4; ++fn) s[fn] = v8f_zero();
    #pragma unroll
    for (int kkk = 0; kkk < 2; ++kkk) {
      #pragma unroll
      for (int fn = 0; fn < 4; ++fn) {
        const u16* p = &Ks[(fn * 16 + nlane) * 64 + kkk * 32 + hiB];
        v16bf b = cat16(*(const v8bf*)p, *(const v8bf*)(p + 16));
        s[fn] = __builtin_amdgcn_wmma_f32_16x16x32_bf16(
            false, qf[kkk], false, b, (short)0, s[fn], false, false);
      }
    }

    // scale + causal mask
    #pragma unroll
    for (int fn = 0; fn < 4; ++fn) {
      int colg = jb * 64 + fn * 16 + nlane;
      #pragma unroll
      for (int r = 0; r < 8; ++r) {
        int rowg = rbase + r + hi8;
        float v = s[fn][r] * scale;
        s[fn][r] = (colg <= rowg) ? v : -1e30f;
      }
    }

    // online softmax (row groups: lanes 0-15 = rows r, lanes 16-31 = rows r+8)
    #pragma unroll
    for (int r = 0; r < 8; ++r) {
      float mloc = -1e30f;
      #pragma unroll
      for (int fn = 0; fn < 4; ++fn) mloc = fmaxf(mloc, s[fn][r]);
      #pragma unroll
      for (int off = 1; off < 16; off <<= 1)
        mloc = fmaxf(mloc, __shfl_xor(mloc, off, 32));
      float mnew  = fmaxf(m_r[r], mloc);
      float alpha = __expf(m_r[r] - mnew);
      float lsum  = 0.0f;
      #pragma unroll
      for (int fn = 0; fn < 4; ++fn) {
        float p = __expf(s[fn][r] - mnew);
        s[fn][r] = p;
        lsum += p;
      }
      #pragma unroll
      for (int off = 1; off < 16; off <<= 1) lsum += __shfl_xor(lsum, off, 32);
      m_r[r] = mnew;
      l_r[r] = l_r[r] * alpha + lsum;
      #pragma unroll
      for (int fd = 0; fd < 4; ++fd) o[fd][r] *= alpha;
    }

    // P: C-layout f32 -> LDS -> A-layout bf16 (wave-private buffer, no barrier needed)
    #pragma unroll
    for (int fn = 0; fn < 4; ++fn)
      #pragma unroll
      for (int r = 0; r < 8; ++r)
        Pw[(r + hi8) * 64 + fn * 16 + nlane] = f2bf(s[fn][r]);

    // O += P * V
    #pragma unroll
    for (int kkk = 0; kkk < 2; ++kkk) {
      const u16* ap = &Pw[nlane * 64 + kkk * 32 + hiA];
      v16bf a = cat16(*(const v8bf*)ap, *(const v8bf*)(ap + 16));
      #pragma unroll
      for (int fd = 0; fd < 4; ++fd) {
        const u16* bp = &Vt[(fd * 16 + nlane) * 64 + kkk * 32 + hiB];
        v16bf b = cat16(*(const v8bf*)bp, *(const v8bf*)(bp + 16));
        o[fd] = __builtin_amdgcn_wmma_f32_16x16x32_bf16(
            false, a, false, b, (short)0, o[fd], false, false);
      }
    }
  }

  // normalize and store y[b][row][h*64+d] (f32)
  #pragma unroll
  for (int r = 0; r < 8; ++r) {
    float inv = 1.0f / l_r[r];
    int rowg = rbase + r + hi8;
    float* op = out + ((size_t)bi * LL + rowg) * CC + h * DD;
    #pragma unroll
    for (int fd = 0; fd < 4; ++fd)
      op[fd * 16 + nlane] = o[fd][r] * inv;
  }
}

extern "C" void kernel_launch(void* const* d_in, const int* in_sizes, int n_in,
                              void* d_out, int out_size, void* d_ws, size_t ws_size,
                              hipStream_t stream) {
  (void)in_sizes; (void)n_in; (void)out_size; (void)ws_size;
  const float* x    = (const float*)d_in[0];
  const float* W    = (const float*)d_in[1];
  const float* bias = (const float*)d_in[2];
  float* out = (float*)d_out;

  // workspace layout (bf16): xb[M*K] | Wt[N*K] | Q | K | V  (~70 MB)
  u16* xb = (u16*)d_ws;
  u16* wt = xb + (size_t)MM * KK;
  u16* qw = wt + (size_t)NN * KK;
  u16* kw = qw + (size_t)BB * HH * LL * DD;
  u16* vw = kw + (size_t)BB * HH * LL * DD;

  int n4 = MM * KK / 4;
  cvt_x_kernel<<<(n4 + 255) / 256, 256, 0, stream>>>((const float4*)x, xb, n4);
  int nw = KK * NN;
  cvt_wT_kernel<<<(nw + 255) / 256, 256, 0, stream>>>(W, wt);
  qkv_gemm_kernel<<<dim3(NN / 128, MM / 128), 256, 0, stream>>>(xb, wt, bias, qw, kw, vw);
  attn_kernel<<<dim3(LL / 128, HH, BB), 256, 0, stream>>>(qw, kw, vw, out);
}
